// QAttention_20461224198681
// MI455X (gfx1250) — 16-bit and 8-bit matrix operand order repaired and checked by reading, compiled, not run on hardware
//
#include <hip/hip_runtime.h>
#include <hip/hip_bf16.h>
#include <cstdint>


#define BB   64
#define NN   197
#define CC   768
#define HH   12
#define DD   64
#define C3   2304
#define MM   12608
#define NT13 13
#define NPAD 208
#define NP32 224
#define BROW 40

typedef _Float16 v16h __attribute__((ext_vector_type(16)));
typedef _Float16 h8   __attribute__((ext_vector_type(8)));
typedef float    v8f  __attribute__((ext_vector_type(8)));
typedef int      v8i  __attribute__((ext_vector_type(8)));

#if __has_builtin(__builtin_amdgcn_global_load_async_to_lds_b128) && \
    __has_builtin(__builtin_amdgcn_s_wait_asynccnt)
#define ASYNC_LDS 1
typedef int b128_t __attribute__((vector_size(16)));
typedef b128_t __attribute__((address_space(1)))* gb128p;
typedef b128_t __attribute__((address_space(3)))* lb128p;
#else
#define ASYNC_LDS 0
#endif

__device__ __forceinline__ void copy16_g2l(const _Float16* __restrict__ g,
                                           _Float16* l) {
#if ASYNC_LDS
    __builtin_amdgcn_global_load_async_to_lds_b128((gb128p)g, (lb128p)l, 0, 0);
#else
    *(h8*)l = *(const h8*)g;
#endif
}

__device__ __forceinline__ void wait_async_le(int pending) {
#if ASYNC_LDS
    if (pending) __builtin_amdgcn_s_wait_asynccnt(2);
    else         __builtin_amdgcn_s_wait_asynccnt(0);
#else
    (void)pending;
#endif
}


__device__ __forceinline__ v16h load_a_f16(const _Float16* __restrict__ row,
                                           int kc, int lane) {
    const int kb = kc + ((lane >> 4) << 4);
    h8 x0 = *(const h8*)(row + kb);
    h8 x1 = *(const h8*)(row + kb + 8);
    v16h a;
#pragma unroll
    for (int i = 0; i < 8; ++i) { a[i] = x0[i]; a[i + 8] = x1[i]; }
    return a;
}

__device__ __forceinline__ v16h load_b_lds(const _Float16* buf, int t,
                                           int lane) {
    const _Float16* p = buf + (t * 16 + (lane & 15)) * BROW + ((lane >> 4) << 4);
    h8 x0 = *(const h8*)p;
    h8 x1 = *(const h8*)(p + 8);
    v16h b;
#pragma unroll
    for (int i = 0; i < 8; ++i) { b[i] = x0[i]; b[i + 8] = x1[i]; }
    return b;
}

__device__ __forceinline__ void stage_b_tile(_Float16* dst,
                                             const _Float16* __restrict__ w,
                                             int n0, int kc, int tid) {
#pragma unroll
    for (int it = 0; it < 2; ++it) {
        const int seg = tid + it * 256;
        const int r = seg >> 2, s = seg & 3;
        copy16_g2l(w + (size_t)(n0 + r) * CC + kc + s * 8,
                   dst + r * BROW + s * 8);
    }
}

__device__ __forceinline__ float fq_clip(float x) {
    return fminf(fmaxf(rintf(x), -128.0f), 127.0f);
}

__global__ void cvt_f16_kernel(const float* __restrict__ x,
                               _Float16* __restrict__ out, int n) {
    for (int i = blockIdx.x * blockDim.x + threadIdx.x; i < n;
         i += gridDim.x * blockDim.x)
        out[i] = (_Float16)x[i];
}

__global__ void quant_w_kernel(const float* __restrict__ w,
                               const float* __restrict__ s_p,
                               _Float16* __restrict__ out, int n) {
    const float inv = 1.0f / s_p[0];
    for (int i = blockIdx.x * blockDim.x + threadIdx.x; i < n;
         i += gridDim.x * blockDim.x)
        out[i] = (_Float16)fq_clip(w[i] * inv);
}

__global__ __launch_bounds__(256) void qkv_gemm_kernel(
    const _Float16* __restrict__ xh, const _Float16* __restrict__ wq,
    const float* __restrict__ bias, const float* __restrict__ wqs_p,
    const float* __restrict__ a1_p, int8_t* __restrict__ q8,
    int8_t* __restrict__ k8, int8_t* __restrict__ v8) {
    __shared__ _Float16 sB[2][128 * BROW];
    const int tid = threadIdx.x, wave = tid >> 5, lane = tid & 31;
    int m0 = blockIdx.x * 128 + wave * 16;
    if (m0 > MM - 16) m0 = MM - 16;
    const int n0 = blockIdx.y * 128;
    const float wqs = wqs_p[0];
    const float inva1 = 1.0f / a1_p[0];

    const _Float16* aRow = xh + (size_t)(m0 + (lane & 15)) * CC;
    v8f acc[8] = {{}, {}, {}, {}, {}, {}, {}, {}};

    stage_b_tile(sB[0], wq, n0, 0, tid);
    int buf = 0;
    for (int kc = 0; kc < CC; kc += 32, buf ^= 1) {
        const bool hasNext = (kc + 32) < CC;
        if (hasNext) stage_b_tile(sB[buf ^ 1], wq, n0, kc + 32, tid);
        wait_async_le(hasNext ? 1 : 0);
        __syncthreads();
        v16h af = load_a_f16(aRow, kc, lane);
        __builtin_prefetch(aRow + kc + 64, 0, 0);
#pragma unroll
        for (int t = 0; t < 8; ++t) {
            v16h bf = load_b_lds(sB[buf], t, lane);
            acc[t] = __builtin_amdgcn_wmma_f32_16x16x32_f16(
                false, af, false, bf, (short)0, acc[t], false, false);
        }
        __syncthreads();
    }

#pragma unroll
    for (int t = 0; t < 8; ++t) {
#pragma unroll
        for (int r = 0; r < 8; ++r) {
            const int m  = m0 + ((lane >> 4) << 3) + r;
            const int n3 = n0 + t * 16 + (lane & 15);
            const float h = acc[t][r] * wqs + bias[n3];
            const float xq = fq_clip(h * inva1);
            const int part = n3 / CC;
            const int c = n3 - part * CC;
            const int hh = c >> 6, d = c & 63;
            const int bi = m / NN, ni = m - bi * NN;
            const size_t off = (((size_t)(bi * HH + hh) * NN + ni) << 6) + d;
            int8_t* dst = (part == 0) ? q8 : (part == 1 ? k8 : v8);
            dst[off] = (int8_t)xq;
        }
    }
}

__global__ __launch_bounds__(128) void attn_kernel(
    const int8_t* __restrict__ q8, const int8_t* __restrict__ k8,
    const int8_t* __restrict__ v8, const float* __restrict__ a1_p,
    const float* __restrict__ attn_p, const float* __restrict__ a2_p,
    _Float16* __restrict__ ctx16) {
    __shared__ float    sS[16][NPAD];
    __shared__ _Float16 sP[16][NP32];
    __shared__ _Float16 sV[NP32][DD];
    __shared__ float    sRed[16][8];
    __shared__ float    sMax[16], sSum[16];

    const int tid = threadIdx.x, wave = tid >> 5, lane = tid & 31;
    int blk = blockIdx.x;
    const int mtile = blk % NT13; blk /= NT13;
    const int h = blk % HH;
    const int b = blk / HH;
    const int m0 = mtile * 16;

    const float a1 = a1_p[0], s = attn_p[0], a2 = a2_p[0];
    const float invS = 1.0f / s;
    const float scaleQK = a1 * a1 * 0.125f;

    const size_t bhBase = (size_t)(b * HH + h) * NN * DD;
    const int8_t* qB = q8 + bhBase;
    const int8_t* kB = k8 + bhBase;
    const int8_t* vB = v8 + bhBase;

    for (int idx = tid; idx < NP32 * DD; idx += 128) {
        const int k = idx >> 6, d = idx & 63;
        sV[k][d] = (k < NN) ? (_Float16)(float)vB[((size_t)k << 6) + d]
                            : (_Float16)0.0f;
    }

    const int mrow = m0 + (lane & 15);
    const bool mval = (mrow < NN);
    const int8_t* qRow = qB + ((size_t)mrow << 6);
    const int kbA = (lane >> 4) << 3;
    v8i afrag;
#pragma unroll
    for (int j = 0; j < 8; ++j) {
        const int off = kbA + ((j >> 1) << 4) + ((j & 1) << 2);
        afrag[j] = mval ? *(const int*)(qRow + off) : 0;
    }

    for (int t = wave; t < NT13; t += 4) {
        const int ncol = t * 16 + (lane & 15);
        const bool nv = (ncol < NN);
        const int8_t* kRow = kB + ((size_t)ncol << 6);
        const int kbB = (lane >> 4) << 4;
        v8i bfrag;
#pragma unroll
        for (int j = 0; j < 8; ++j) {
            const int off = (kbB >> 1) + ((j >> 1) << 4) + ((j & 1) << 2);
            bfrag[j] = nv ? *(const int*)(kRow + off) : 0;
        }
        v8i acc = {};
        acc = __builtin_amdgcn_wmma_i32_16x16x64_iu8(
            true, afrag, true, bfrag, acc, false, false);
#pragma unroll
        for (int r = 0; r < 8; ++r) {
            const int lm = ((lane >> 4) << 3) + r;
            const float attn = (float)acc[r] * scaleQK;
            sS[lm][t * 16 + (lane & 15)] = fq_clip(attn * invS);
        }
    }
    __syncthreads();

    {
        const int r = tid >> 3, cc = tid & 7;
        float mx = -1e30f;
        for (int n = cc; n < NN; n += 8) mx = fmaxf(mx, sS[r][n]);
        sRed[r][cc] = mx;
    }
    __syncthreads();
    if (tid < 16) {
        float mx = sRed[tid][0];
#pragma unroll
        for (int j = 1; j < 8; ++j) mx = fmaxf(mx, sRed[tid][j]);
        sMax[tid] = mx;
    }
    __syncthreads();

    const float x0i  = floorf(-0.6931f * invS);
    const float bInt = floorf((0.96963238f / 0.35815147f) * invS);
    const float cInt = floorf((1.0f / 0.35815147f) * invS * invS);
    for (int idx = tid; idx < 16 * NN; idx += 128) {
        const int r = idx / NN, n = idx - r * NN;
        float xi = sS[r][n] - sMax[r];
        xi = fmaxf(xi, 30.0f * x0i);
        const float qf = floorf(xi / x0i);
        const float rr = xi - x0i * qf;
        const float z  = rr * (rr + bInt) + cInt;
        sS[r][n] = fmaxf(floorf(z * exp2f(30.0f - qf)), 0.0f);
    }
    __syncthreads();

    {
        const int r = tid >> 3, cc = tid & 7;
        float sm = 0.0f;
        for (int n = cc; n < NN; n += 8) sm += sS[r][n];
        sRed[r][cc] = sm;
    }
    __syncthreads();
    if (tid < 16) {
        float sm = 0.0f;
#pragma unroll
        for (int j = 0; j < 8; ++j) sm += sRed[tid][j];
        sSum[tid] = sm;
    }
    __syncthreads();

    for (int idx = tid; idx < 16 * NP32; idx += 128) {
        const int r = idx / NP32, n = idx - r * NP32;
        float p = 0.0f;
        if (n < NN) {
            const float e  = sS[r][n];
            const float so = rintf(sSum[r] / e);
            const float big = floorf(log2f(so));
            const float rounds =
                (so - exp2f(big) >= exp2f(big - 1.0f)) ? big + 1.0f : big;
            const float ql = fminf(fmaxf(rounds, 0.0f), 15.0f);
            p = (rounds >= 16.0f) ? 0.0f : exp2f(-ql);
        }
        sP[r][n] = (_Float16)p;
    }
    __syncthreads();

    {
        const int dt = wave;
        v8f acc = {};
        for (int kc = 0; kc < NP32; kc += 32) {
            const int rA   = lane & 15;
            const int kbA2 = kc + ((lane >> 4) << 4);
            v16h af;
#pragma unroll
            for (int i = 0; i < 8; ++i) {
                af[i]     = sP[rA][kbA2 + i];
                af[i + 8] = sP[rA][kbA2 + 8 + i];
            }
            const int col  = dt * 16 + (lane & 15);
            const int kbB2 = kc + ((lane >> 4) << 4);
            v16h bf;
#pragma unroll
            for (int j = 0; j < 16; ++j) bf[j] = sV[kbB2 + j][col];
            acc = __builtin_amdgcn_wmma_f32_16x16x32_f16(
                false, af, false, bf, (short)0, acc, false, false);
        }
#pragma unroll
        for (int r = 0; r < 8; ++r) {
            const int m = m0 + ((lane >> 4) << 3) + r;
            if (m < NN) {
                const int d = dt * 16 + (lane & 15);
                const float ctx = acc[r] * a1;
                ctx16[((size_t)(b * NN + m)) * CC + h * DD + d] =
                    (_Float16)fq_clip(ctx / a2);
            }
        }
    }
}

__global__ __launch_bounds__(256) void proj_gemm_kernel(
    const _Float16* __restrict__ ctx16, const _Float16* __restrict__ wp,
    const float* __restrict__ bias, const float* __restrict__ wps_p,
    const float* __restrict__ a2_p, const float* __restrict__ a3_p,
    float* __restrict__ out) {
    __shared__ _Float16 sB[2][128 * BROW];
    const int tid = threadIdx.x, wave = tid >> 5, lane = tid & 31;
    int m0 = blockIdx.x * 128 + wave * 16;
    if (m0 > MM - 16) m0 = MM - 16;
    const int n0 = blockIdx.y * 128;
    const float scl = a2_p[0] * wps_p[0];
    const float a3 = a3_p[0];
    const float inva3 = 1.0f / a3;

    const _Float16* aRow = ctx16 + (size_t)(m0 + (lane & 15)) * CC;
    v8f acc[8] = {{}, {}, {}, {}, {}, {}, {}, {}};

    stage_b_tile(sB[0], wp, n0, 0, tid);
    int buf = 0;
    for (int kc = 0; kc < CC; kc += 32, buf ^= 1) {
        const bool hasNext = (kc + 32) < CC;
        if (hasNext) stage_b_tile(sB[buf ^ 1], wp, n0, kc + 32, tid);
        wait_async_le(hasNext ? 1 : 0);
        __syncthreads();
        v16h af = load_a_f16(aRow, kc, lane);
        __builtin_prefetch(aRow + kc + 64, 0, 0);
#pragma unroll
        for (int t = 0; t < 8; ++t) {
            v16h bf = load_b_lds(sB[buf], t, lane);
            acc[t] = __builtin_amdgcn_wmma_f32_16x16x32_f16(
                false, af, false, bf, (short)0, acc[t], false, false);
        }
        __syncthreads();
    }

#pragma unroll
    for (int t = 0; t < 8; ++t) {
#pragma unroll
        for (int r = 0; r < 8; ++r) {
            const int m = m0 + ((lane >> 4) << 3) + r;
            const int n = n0 + t * 16 + (lane & 15);
            const float v = acc[t][r] * scl + bias[n];
            out[(size_t)m * CC + n] = fq_clip(v * inva3) * a3;
        }
    }
}

static inline size_t alignup(size_t x) { return (x + 255) & ~(size_t)255; }

extern "C" void kernel_launch(void* const* d_in, const int* in_sizes, int n_in,
                              void* d_out, int out_size, void* d_ws,
                              size_t ws_size, hipStream_t stream) {
    (void)in_sizes; (void)n_in; (void)out_size; (void)ws_size;
    const float* x        = (const float*)d_in[0];
    const float* qkv_w    = (const float*)d_in[1];
    const float* qkv_b    = (const float*)d_in[2];
    const float* proj_w   = (const float*)d_in[3];
    const float* proj_b   = (const float*)d_in[4];
    const float* w_qkv_s  = (const float*)d_in[5];
    const float* a1_s     = (const float*)d_in[6];
    const float* attn_s   = (const float*)d_in[7];
    const float* a2_s     = (const float*)d_in[8];
    const float* w_proj_s = (const float*)d_in[9];
    const float* a3_s     = (const float*)d_in[10];

    char* ws = (char*)d_ws;
    size_t off = 0;
    _Float16* xh   = (_Float16*)(ws + off); off += alignup((size_t)MM * CC * 2);
    _Float16* wq16 = (_Float16*)(ws + off); off += alignup((size_t)C3 * CC * 2);
    _Float16* wp16 = (_Float16*)(ws + off); off += alignup((size_t)CC * CC * 2);
    int8_t*   q8   = (int8_t*)  (ws + off); off += alignup((size_t)BB * HH * NN * DD);
    int8_t*   k8   = (int8_t*)  (ws + off); off += alignup((size_t)BB * HH * NN * DD);
    int8_t*   v8   = (int8_t*)  (ws + off); off += alignup((size_t)BB * HH * NN * DD);
    _Float16* ctx16= (_Float16*)(ws + off); off += alignup((size_t)MM * CC * 2);

    cvt_f16_kernel<<<2048, 256, 0, stream>>>(x, xh, MM * CC);
    quant_w_kernel<<<2048, 256, 0, stream>>>(qkv_w, w_qkv_s, wq16, C3 * CC);
    quant_w_kernel<<<1024, 256, 0, stream>>>(proj_w, w_proj_s, wp16, CC * CC);

    qkv_gemm_kernel<<<dim3((MM + 127) / 128, C3 / 128), 256, 0, stream>>>(
        xh, wq16, qkv_b, w_qkv_s, a1_s, q8, k8, v8);

    attn_kernel<<<BB * HH * NT13, 128, 0, stream>>>(
        q8, k8, v8, a1_s, attn_s, a2_s, ctx16);

    proj_gemm_kernel<<<dim3((MM + 127) / 128, CC / 128), 256, 0, stream>>>(
        ctx16, wp16, proj_b, w_proj_s, a2_s, a3_s, (float*)d_out);
}
